// HybridMethod_53085795778708
// MI455X (gfx1250) — matrix operand order repaired and checked by reading, not compiled, not run on hardware
//
#include <hip/hip_runtime.h>
#include <hip/hip_bf16.h>

// ---------------------------------------------------------------------------
// GNN forward for MI455X (gfx1250, wave32, WMMA).
// One tiled bf16-WMMA GEMM kernel powers every matmul:
//   C[M,N] (f32) = opt_relu( A[M,K] (f32, cvt->bf16 inline) @ B[K,N] (f32, cvt) + bias )
// ---------------------------------------------------------------------------

typedef __attribute__((ext_vector_type(16))) __bf16 v16bf;
typedef __attribute__((ext_vector_type(2)))  __bf16 v2bf;
typedef __attribute__((ext_vector_type(8)))  float  v8f;

#define BM 128
#define BN 128
#define BK 32
#define NTHREADS 256

// two fp32 -> packed bf16x2 (lo = a, hi = b).
// Fast path: v_cvt_pk_bf16_f32 if the toolchain has it.
// Fallback: round-half-up (+0x8000) then one v_perm_b32 to merge high halves:
//   dst bytes = { ua[2], ua[3], ub[2], ub[3] }  -> sel 0x07060302 with (src0=ub, src1=ua)
__device__ __forceinline__ unsigned pk_bf16(float a, float b) {
#if defined(__has_builtin) && __has_builtin(__builtin_amdgcn_cvt_pk_bf16_f32)
  v2bf r = __builtin_amdgcn_cvt_pk_bf16_f32(a, b);
  return __builtin_bit_cast(unsigned, r);
#else
  unsigned ua = __float_as_uint(a) + 0x8000u;
  unsigned ub = __float_as_uint(b) + 0x8000u;
  return __builtin_amdgcn_perm(ub, ua, 0x07060302u);
#endif
}

// LDS fragment layout (per 16-row subtile), contiguous k order (k = 16*(lane>>4) + idx), same for A and B:
//  A (16x32 bf16):  lane = (m&15) + 16*(k>>4)      ; idx = (k&15)
//  B (32x16 bf16):  lane = (n&15) + 16*(k>>4)      ; idx = (k&15)
//  ushort offset    = subtile*512 + lane*16 + idx   (32 contiguous bytes/lane)
__global__ __launch_bounds__(NTHREADS)
void gemm_bf16_wmma(const float* __restrict__ A, const float* __restrict__ B,
                    const float* __restrict__ bias, float* __restrict__ C,
                    int K, int lda, int ldb, int ldc, int relu)
{
  __shared__ __align__(32) unsigned short smem[2][8192]; // [buf][ A:0..4095 | B:4096..8191 ]

  const int t    = threadIdx.x;
  const int lane = t & 31;
  const int wid  = t >> 5;
  const int wm   = wid >> 1;   // 0..3 : wave row  (32 rows each)
  const int wn   = wid & 1;    // 0..1 : wave col  (64 cols each)
  const long blockM = (long)blockIdx.y * BM;
  const long blockN = (long)blockIdx.x * BN;

  // --- A staging indices: 4 float4 per thread over a 128x32 tile ---
  int aRow[4], aKq[4], aOff[4];
  #pragma unroll
  for (int i = 0; i < 4; ++i) {
    int f = t + i * NTHREADS;          // float4 slot id, 0..1023
    aRow[i] = f >> 3;                  // 128 rows x 8 float4 along K
    aKq[i]  = (f & 7) << 2;
    int halfSel = aKq[i] >> 4;
    int idx     = aKq[i] & 15;
    aOff[i] = ((aRow[i] >> 4) << 9) + (((aRow[i] & 15) + (halfSel << 4)) << 4) + idx;
  }
  // --- B staging indices: 2(k) x 8(n) patch per thread over a 32x128 tile ---
  const int bk  = (t >> 4) << 1;       // even k row, 0..30
  const int bNo = (t & 15) << 3;       // n octet base, 0..120
  const int bKhalf = bk >> 4;

  float4 aReg[4], bk0[2], bk1[2];

  auto loadRegs = [&](int k0) {
    #pragma unroll
    for (int i = 0; i < 4; ++i)
      aReg[i] = *(const float4*)(A + (blockM + aRow[i]) * (long)lda + k0 + aKq[i]);
    const float* brow0 = B + (long)(k0 + bk) * ldb + blockN + bNo;
    const float* brow1 = brow0 + ldb;
    bk0[0] = *(const float4*)(brow0);
    bk0[1] = *(const float4*)(brow0 + 4);
    bk1[0] = *(const float4*)(brow1);
    bk1[1] = *(const float4*)(brow1 + 4);
  };

  auto storeLds = [&](int buf) {
    unsigned short* sA = &smem[buf][0];
    unsigned short* sB = &smem[buf][4096];
    #pragma unroll
    for (int i = 0; i < 4; ++i) {
      unsigned lo = pk_bf16(aReg[i].x, aReg[i].y);
      unsigned hi = pk_bf16(aReg[i].z, aReg[i].w);
      *(uint2*)(&sA[aOff[i]]) = make_uint2(lo, hi);        // 8B frag-layout store
    }
    #pragma unroll
    for (int q = 0; q < 2; ++q) {
      float a0[4] = {bk0[q].x, bk0[q].y, bk0[q].z, bk0[q].w};
      float a1[4] = {bk1[q].x, bk1[q].y, bk1[q].z, bk1[q].w};
      #pragma unroll
      for (int i = 0; i < 4; ++i) {
        int n = bNo + (q << 2) + i;
        int off = ((n >> 4) << 9) + (((n & 15) + (bKhalf << 4)) << 4) + (bk & 15);
        *(unsigned*)(&sB[off]) = pk_bf16(a0[i], a1[i]);    // 4B store: (k, k+1)
      }
    }
  };

  v8f acc[2][4];
  #pragma unroll
  for (int i = 0; i < 2; ++i)
    #pragma unroll
    for (int j = 0; j < 4; ++j)
      acc[i][j] = v8f{0.f,0.f,0.f,0.f,0.f,0.f,0.f,0.f};

  const int nk = K / BK;
  loadRegs(0);
  storeLds(0);

  for (int kb = 0; kb < nk; ++kb) {
    const bool hasNext = (kb + 1) < nk;
    if (hasNext) loadRegs((kb + 1) * BK);        // global loads for next tile
    if (kb + 3 < nk)                             // prefetch the big A (adj) stream ahead
      __builtin_prefetch(A + (blockM + aRow[0]) * (long)lda + (kb + 3) * BK + aKq[0], 0, 1);
    __syncthreads();                             // current buf LDS visible
    {
      const int buf = kb & 1;
      const v16bf* pA = (const v16bf*)&smem[buf][0];
      const v16bf* pB = (const v16bf*)&smem[buf][4096];
      v16bf af[2], bfr[4];
      #pragma unroll
      for (int i = 0; i < 2; ++i) af[i]  = pA[((wm * 2 + i) << 5) + lane];
      #pragma unroll
      for (int j = 0; j < 4; ++j) bfr[j] = pB[((wn * 4 + j) << 5) + lane];
      #pragma unroll
      for (int i = 0; i < 2; ++i)
        #pragma unroll
        for (int j = 0; j < 4; ++j)
          acc[i][j] = __builtin_amdgcn_wmma_f32_16x16x32_bf16(
              false, af[i], false, bfr[j], (short)0, acc[i][j], false, false);
    }
    if (hasNext) storeLds((kb + 1) & 1);         // fill other buf (consumed pre-barrier)
  }

  // --- epilogue: C layout is (m = v + 8*(lane>>4), n = lane&15) per 16x16 tile ---
  #pragma unroll
  for (int i = 0; i < 2; ++i) {
    const long row0 = blockM + (long)(wm * 2 + i) * 16 + ((lane >> 4) << 3);
    #pragma unroll
    for (int j = 0; j < 4; ++j) {
      const long col = blockN + (long)(wn * 4 + j) * 16 + (lane & 15);
      const float bs = bias ? bias[col] : 0.0f;
      #pragma unroll
      for (int v = 0; v < 8; ++v) {
        float val = acc[i][j][v] + bs;
        if (relu) val = fmaxf(val, 0.0f);
        C[(row0 + v) * (long)ldc + col] = val;
      }
    }
  }
}

// cat[:, 256:768] = target_feats   (float4-wide copy)
__global__ __launch_bounds__(256)
void pack_cat(const float* __restrict__ tf, float* __restrict__ cat) {
  long idx = (long)blockIdx.x * 256 + threadIdx.x;   // 8192*128 float4 slots
  long r = idx >> 7, c = (idx & 127) << 2;
  *(float4*)(cat + r * 768 + 256 + c) = *(const float4*)(tf + r * 512 + c);
}

// row-wise log_softmax over 512 columns, one block (128 thr) per row
__global__ __launch_bounds__(128)
void log_softmax_512(const float* __restrict__ X, float* __restrict__ out) {
  __shared__ float red[128];
  const int t = threadIdx.x;
  const long r = blockIdx.x;
  const float* row = X + r * 512;
  float v[4];
  float m = -3.402823466e38f;
  #pragma unroll
  for (int i = 0; i < 4; ++i) { v[i] = row[t + i * 128]; m = fmaxf(m, v[i]); }
  red[t] = m; __syncthreads();
  for (int s = 64; s > 0; s >>= 1) { if (t < s) red[t] = fmaxf(red[t], red[t + s]); __syncthreads(); }
  m = red[0]; __syncthreads();
  float sum = 0.f;
  #pragma unroll
  for (int i = 0; i < 4; ++i) sum += __expf(v[i] - m);
  red[t] = sum; __syncthreads();
  for (int s = 64; s > 0; s >>= 1) { if (t < s) red[t] += red[t + s]; __syncthreads(); }
  const float lse = m + __logf(red[0]);
  float* orow = out + r * 512;
  #pragma unroll
  for (int i = 0; i < 4; ++i) orow[t + i * 128] = v[i] - lse;
}

extern "C" void kernel_launch(void* const* d_in, const int* in_sizes, int n_in,
                              void* d_out, int out_size, void* d_ws, size_t ws_size,
                              hipStream_t stream) {
  (void)in_sizes; (void)n_in; (void)out_size; (void)ws_size;
  const float* x    = (const float*)d_in[0];
  const float* tf   = (const float*)d_in[1];
  const float* adj0 = (const float*)d_in[2];
  const float* adj1 = (const float*)d_in[3];
  const float* w1   = (const float*)d_in[4];
  const float* b1   = (const float*)d_in[5];
  const float* w2   = (const float*)d_in[6];
  const float* b2   = (const float*)d_in[7];
  const float* lw1  = (const float*)d_in[8];
  const float* lb1  = (const float*)d_in[9];
  const float* lw2  = (const float*)d_in[10];
  const float* lb2  = (const float*)d_in[11];
  const float* lw3  = (const float*)d_in[12];
  const float* lb3  = (const float*)d_in[13];
  float* out = (float*)d_out;

  char* ws = (char*)d_ws;
  float* XW1 = (float*)(ws);                         //  8 MB: 8192x256  (reused for HW2)
  float* H1  = (float*)(ws + (8ul  << 20));          //  8 MB: 8192x256
  float* CAT = (float*)(ws + (16ul << 20));          // 24 MB: 8192x768  (reused for O5)
  float* O4  = (float*)(ws + (40ul << 20));          // 24 MB: 8192x768  (reused for O6)

  const dim3 blk(NTHREADS);
  const int MB = 8192 / BM;                          // 64 row-blocks everywhere

  // 1) XW1 = x @ gc1_w                       [8192,512]x[512,256]
  gemm_bf16_wmma<<<dim3(256 / BN, MB), blk, 0, stream>>>(x, w1, nullptr, XW1, 512, 512, 256, 256, 0);
  // 2) H1 = relu(adj0 @ XW1 + gc1_b)         [8192,8192]x[8192,256]
  gemm_bf16_wmma<<<dim3(256 / BN, MB), blk, 0, stream>>>(adj0, XW1, b1, H1, 8192, 8192, 256, 256, 1);
  // 3) HW2 = H1 @ gc2_w  (-> XW1 buffer)     [8192,256]x[256,256]
  gemm_bf16_wmma<<<dim3(256 / BN, MB), blk, 0, stream>>>(H1, w2, nullptr, XW1, 256, 256, 256, 256, 0);
  // 4) H2 = adj1 @ HW2 + gc2_b  -> CAT[:, 0:256]  (ldc = 768)
  gemm_bf16_wmma<<<dim3(256 / BN, MB), blk, 0, stream>>>(adj1, XW1, b2, CAT, 8192, 8192, 256, 768, 0);
  // 5) CAT[:, 256:768] = target_feats
  pack_cat<<<dim3((8192 * 128) / 256), dim3(256), 0, stream>>>(tf, CAT);
  // 6) O4 = relu(CAT @ lin1_w + lin1_b)      [8192,768]x[768,768]
  gemm_bf16_wmma<<<dim3(768 / BN, MB), blk, 0, stream>>>(CAT, lw1, lb1, O4, 768, 768, 768, 768, 1);
  // 7) O5 = relu(O4 @ lin2_w + lin2_b)  -> CAT buffer
  gemm_bf16_wmma<<<dim3(768 / BN, MB), blk, 0, stream>>>(O4, lw2, lb2, CAT, 768, 768, 768, 768, 1);
  // 8) O6 = O5 @ lin3_w + lin3_b  -> O4 buffer    [8192,768]x[768,512]
  gemm_bf16_wmma<<<dim3(512 / BN, MB), blk, 0, stream>>>(CAT, lw3, lb3, O4, 768, 768, 512, 512, 0);
  // 9) out = log_softmax(O6, axis=1)
  log_softmax_512<<<dim3(8192), dim3(128), 0, stream>>>(O4, out);
}
